// CAPERNNAgent_60206851555654
// MI455X (gfx1250) — matrix operand order repaired and checked by reading, not compiled, not run on hardware
//
#include <hip/hip_runtime.h>
#include <math.h>

// ---------------------------------------------------------------------------
// CAPE RNN agent, fully fused for MI455X (gfx1250, wave32, WMMA f16->f32).
// One workgroup = 16 batch rows, 256 threads (8 waves), ~307KB dynamic LDS.
// Weights are pre-packed (f32 -> f16) into WMMA-B-fragment-native layout in
// d_ws by a small pack kernel; the fused kernel streams them from L2.
// ---------------------------------------------------------------------------

typedef _Float16 h16;
typedef __attribute__((ext_vector_type(8)))  _Float16 v8h;
typedef __attribute__((ext_vector_type(16))) _Float16 v16h;
typedef __attribute__((ext_vector_type(8)))  float    v8f;

#define NMAT   26
#define BN_TOT 16384

// ----- LDS layout (bytes) ---------------------------------------------------
#define OFF_EN    0        // f16 enemy_emb  [15][16][192]  = 92160 B (persistent)
#define OFF_AL    92160    // f16 ally_emb   [14][16][192]  = 86016 B
#define OFF_SELF  178176   // f16 self_emb   [16][192]      =  6144 B
#define OFF_X     184320   // f16 staging    12288 halves   = 24576 B
#define OFF_F32A  208896   // f32 scratch A  12288 floats   = 49152 B (q/oBuf/pA/pE/pooled/gi)
#define OFF_F32B  258048   // f32 scratch B  12288 floats   = 49152 B (cA/cE/h0/gh/z)
#define OFF_MASK  307200   // f32 masks: ally [0..223], enemy [224..463]
#define OFF_SC    309248   // f32 scores [b][h][m] / logits [b][m]  (1024 f32)
#define OFF_R     313344   // f32 r softmax [16][4]
#define OFF_QM    313600   // f32 Q_move raw [16][6]
#define SMEM_BYTES 314112

enum { M_SELF1=0,M_SELF2,M_ALLY1,M_ALLY2,M_EN1,M_EN2,
       M_AQ,M_AK,M_AV,M_AO, M_EQ,M_EK,M_EV,M_EO,
       M_FU1,M_FU2,M_PO1,M_PO2,M_RO1,M_RO2,
       M_WIH,M_WHH,M_TR1,M_TR2,M_QP,M_KP };

// ---------------------------------------------------------------------------
// Fragment helpers.
// A (16x32 f16, MxK): lane = m + 16*kh; element e covers
//   K = k0 + kh*16 + e  -> one contiguous 16-half chunk (matches B).
// B (32x16 f16, KxN), same k order as A (this code's choice): lane = n + 16*kh; element e
//   covers K = k0 + kh*16 + e  -> one contiguous 16-half chunk (pack-native).
// C/D (16x16 f32): VGPR v, row M = v + 8*kh, col N = lane&15.
// ---------------------------------------------------------------------------
__device__ __forceinline__ v16h load_a_frag(const h16* A, int SA, int k0, int lane) {
  int m = lane & 15, kh = (lane >> 4) & 1;
  const v8h* p = (const v8h*)(A + (size_t)m * SA + k0 + kh * 16);
  v8h lo = p[0];
  v8h hi = p[1];                       // +8 halves
  return __builtin_shufflevector(lo, hi, 0,1,2,3,4,5,6,7,8,9,10,11,12,13,14,15);
}

__device__ __forceinline__ v16h load_b_frag(const h16* wp, unsigned off, int KB,
                                            int nt, int kb, int lane) {
  const v8h* p = (const v8h*)(wp + off + ((size_t)(nt * KB + kb) * 32u + (unsigned)lane) * 16u);
  v8h lo = p[0], hi = p[1];
  return __builtin_shufflevector(lo, hi, 0,1,2,3,4,5,6,7,8,9,10,11,12,13,14,15);
}

// One 16x16 output tile, loop over K blocks.
__device__ __forceinline__ v8f gemm_one_tile(const h16* A, int SA, const h16* wp,
                                             unsigned off, int KB, int nt, int lane) {
  v8f c = {};
  for (int kb = 0; kb < KB; ++kb) {
    v16h a = load_a_frag(A, SA, kb * 32, lane);
    v16h b = load_b_frag(wp, off, KB, nt, kb, lane);
    c = __builtin_amdgcn_wmma_f32_16x16x32_f16(false, a, false, b, (short)0, c, false, false);
  }
  return c;
}

// 16xN GEMM for one 16-row A tile with A-fragment reuse across NT output tiles
// (NT v8f accumulators stay live in VGPRs: NT=12 -> 96 VGPRs/lane).
template<int NT>
__device__ __forceinline__ void gemm_rowtile(const h16* A, int SA, const h16* wp,
                                             unsigned off, int KB, int lane, v8f* acc) {
  for (int kb = 0; kb < KB; ++kb) {
    v16h a = load_a_frag(A, SA, kb * 32, lane);
#pragma unroll
    for (int t = 0; t < NT; ++t) {
      v16h b = load_b_frag(wp, off, KB, t, kb, lane);
      acc[t] = __builtin_amdgcn_wmma_f32_16x16x32_f16(false, a, false, b, (short)0, acc[t], false, false);
    }
  }
}

__device__ __forceinline__ float geluf(float x) {
  return 0.5f * x * (1.f + erff(x * 0.70710678118654752f));
}
__device__ __forceinline__ float sigmoidf_(float x) { return 1.f / (1.f + expf(-x)); }

__device__ __forceinline__ void store_tile_f32(float* O, int SO, int nt, const v8f& c,
                                               int lane, const float* bias) {
  int kh = (lane >> 4) & 1, cl = lane & 15, col = nt * 16 + cl;
  float bb = bias ? bias[col] : 0.f;
#pragma unroll
  for (int v = 0; v < 8; ++v) O[(size_t)(v + 8 * kh) * SO + col] = c[v] + bb;
}
__device__ __forceinline__ void store_tile_f16(h16* O, int SO, int nt, const v8f& c,
                                               int lane, const float* bias) {
  int kh = (lane >> 4) & 1, cl = lane & 15, col = nt * 16 + cl;
  float bb = bias[col];
#pragma unroll
  for (int v = 0; v < 8; ++v) O[(size_t)(v + 8 * kh) * SO + col] = (h16)(c[v] + bb);
}
__device__ __forceinline__ void store_tile_f16_gelu(h16* O, int SO, int nt, const v8f& c,
                                                    int lane, const float* bias) {
  int kh = (lane >> 4) & 1, cl = lane & 15, col = nt * 16 + cl;
  float bb = bias[col];
#pragma unroll
  for (int v = 0; v < 8; ++v) O[(size_t)(v + 8 * kh) * SO + col] = (h16)geluf(c[v] + bb);
}

// LayerNorm rows of f16 data in LDS (stats in f32, wave32 shfl reduction).
__device__ __forceinline__ void ln_rows_f16(h16* X, int nrows, int W, int SA,
                                            const float* g, const float* bt,
                                            int wave, int lane) {
  for (int r = wave; r < nrows; r += 8) {
    h16* row = X + (size_t)r * SA;
    float s = 0.f, s2 = 0.f;
    for (int j = lane; j < W; j += 32) { float x = (float)row[j]; s += x; s2 += x * x; }
#pragma unroll
    for (int o = 16; o > 0; o >>= 1) { s += __shfl_xor(s, o, 32); s2 += __shfl_xor(s2, o, 32); }
    float m = s / W, vv = s2 / W - m * m;
    float inv = rsqrtf(vv + 1e-5f);
    for (int j = lane; j < W; j += 32) {
      float x = (float)row[j];
      row[j] = (h16)(((x - m) * inv) * g[j] + bt[j]);
    }
  }
}

// ---------------------------------------------------------------------------
// Weight pack kernel: f32 row-major W[K][N] -> f16 B-fragments.
// Fragment (nt,kb): 32 lanes x 16 halves, element e -> K = kb*32 + kh*16 + e,
// N = nt*16 + (lane&15). OOB K/N -> 0 (so K-padding is mathematically inert).
// ---------------------------------------------------------------------------
struct PackArgs {
  const float* src[NMAT];
  int K[NMAT], N[NMAT];
  unsigned off[NMAT];      // in halves
  unsigned cum[NMAT + 1];  // cumulative fragment counts
  h16* dst;
};

__global__ void capernn_pack_weights(PackArgs a) {
  int frag = blockIdx.x * 8 + (threadIdx.x >> 5);
  int lane = threadIdx.x & 31;
  if (frag >= (int)a.cum[NMAT]) return;
  int mi = 0;
  while (frag >= (int)a.cum[mi + 1]) ++mi;
  int lf = frag - (int)a.cum[mi];
  int K = a.K[mi], N = a.N[mi];
  int KB = (K + 31) >> 5;
  int nt = lf / KB, kb = lf % KB;
  const float* W = a.src[mi];
  h16* d = a.dst + a.off[mi] + ((size_t)lf * 32u + (unsigned)lane) * 16u;
  int kh = (lane >> 4) & 1, cl = lane & 15;
  int n = nt * 16 + cl;
#pragma unroll
  for (int e = 0; e < 16; ++e) {
    int k = kb * 32 + kh * 16 + e;
    float v = (k < K && n < N) ? W[(size_t)k * N + n] : 0.f;
    d[e] = (h16)v;
  }
}

// ---------------------------------------------------------------------------
// Fused forward kernel.
// ---------------------------------------------------------------------------
struct KParams {
  const float *own, *allies, *enemies, *hidden, *type_emb;
  const float *b_self1,*g_self1,*bt_self1,*b_self2;
  const float *b_ally1,*g_ally1,*bt_ally1,*b_ally2;
  const float *b_en1,*g_en1,*bt_en1,*b_en2;
  const float *bq_a,*bk_a,*bv_a,*bo_a;
  const float *bq_e,*bk_e,*bv_e,*bo_e;
  const float *b_fu1,*g_fu1,*bt_fu1,*b_fu2;
  const float *b_po1,*g_po1,*bt_po1,*b_po2;
  const float *b_ro1,*g_ro1,*bt_ro1,*b_ro2;
  const float *pre_g,*pre_b,*gbih,*gbhh,*post_g,*post_b;
  const float *b_tr1,*g_tr1,*bt_tr1,*b_tr2;
  const float *mv_W,*mv_b;
  const h16* wp;
  unsigned off[NMAT];
  float *outQ, *outH;
};

__global__ void __launch_bounds__(256) capernn_fused(KParams p) {
  extern __shared__ char smem[];
  h16*   sEN   = (h16*)(smem + OFF_EN);
  h16*   sAL   = (h16*)(smem + OFF_AL);
  h16*   sSELF = (h16*)(smem + OFF_SELF);
  h16*   sX    = (h16*)(smem + OFF_X);   // 12288 halves staging
  h16*   sX2   = sX + 9216;              // secondary staging (<=16x192)
  h16*   sHID  = sX + 4096;              // f16 hidden state (16x256)
  float* fA    = (float*)(smem + OFF_F32A);
  float* fB    = (float*)(smem + OFF_F32B);
  float* sMask = (float*)(smem + OFF_MASK);
  float* sSc   = (float*)(smem + OFF_SC);
  float* sR    = (float*)(smem + OFF_R);
  float* sQM   = (float*)(smem + OFF_QM);
  const h16* wp = p.wp;

  const int tid = threadIdx.x, lane = tid & 31, wave = tid >> 5;
  const int kh = (lane >> 4) & 1, cl = lane & 15;
  const int b0 = blockIdx.x * 16;

  // Warm L2 with the GRU weights (largest matrices) while encoders run.
  if (blockIdx.x == 0) {
    const char* w0 = (const char*)(wp + p.off[M_WIH]);
    size_t nb = (size_t)(p.off[M_TR1] - p.off[M_WIH]) * 2u;  // Wih + Whh bytes
    for (size_t i = (size_t)tid * 256u; i < nb; i += 65536u)
      __builtin_prefetch(w0 + i, 0, 1);
  }

  // ===== Phase E1: stage ally + self encoder inputs (f16, stride 48) =======
  for (int i = tid; i < 12288; i += 256) sX[i] = (h16)0.f;
  __syncthreads();
  if (tid < 240) {
    int ent = tid >> 4, b = tid & 15;
    h16* dst = sX + (size_t)tid * 48;
    const float* src; int F;
    if (ent < 14) { src = p.allies + ((size_t)(b0 + b) * 14 + ent) * 40; F = 40; }
    else          { src = p.own + (size_t)(b0 + b) * 32;                 F = 32; }
    float mabs = 0.f;
    for (int j = 0; j < F; ++j) { float x = src[j]; dst[j] = (h16)x; mabs += fabsf(x); }
    int am = 0; float best = src[F - 9];
    for (int k = 1; k < 9; ++k) { float x = src[F - 9 + k]; if (x > best) { best = x; am = k; } }
    for (int k = 0; k < 8; ++k) dst[F + k] = (h16)p.type_emb[am * 8 + k];
    if (ent < 14) sMask[ent * 16 + b] = (mabs > 0.f) ? 1.f : 0.f;
  }
  __syncthreads();

  // GEMM1 ally/self: K=48/40 (KB=2, zero-padded), N=192. gelu -> f16.
  for (int e = wave; e < 15; e += 8) {
    v8f acc[12] = {};
    const h16* A = sX + (size_t)e * 16 * 48;
    gemm_rowtile<12>(A, 48, wp, (e < 14) ? p.off[M_ALLY1] : p.off[M_SELF1], 2, lane, acc);
    const float* b1 = (e < 14) ? p.b_ally1 : p.b_self1;
    h16* dst = (e < 14) ? (sAL + (size_t)e * 3072) : sSELF;
#pragma unroll
    for (int t = 0; t < 12; ++t) store_tile_f16_gelu(dst, 192, t, acc[t], lane, b1);
  }
  __syncthreads();
  ln_rows_f16(sAL, 224, 192, 192, p.g_ally1, p.bt_ally1, wave, lane);
  ln_rows_f16(sSELF, 16, 192, 192, p.g_self1, p.bt_self1, wave, lane);
  __syncthreads();
  // GEMM2 ally/self (in-place, per-wave entity; DS ops in-order per wave).
  for (int e = wave; e < 15; e += 8) {
    h16* buf = (e < 14) ? (sAL + (size_t)e * 3072) : sSELF;
    v8f acc[12] = {};
    gemm_rowtile<12>(buf, 192, wp, (e < 14) ? p.off[M_ALLY2] : p.off[M_SELF2], 6, lane, acc);
    const float* b2 = (e < 14) ? p.b_ally2 : p.b_self2;
#pragma unroll
    for (int t = 0; t < 12; ++t) store_tile_f16(buf, 192, t, acc[t], lane, b2);
  }
  __syncthreads();

  // ===== Phase E2: enemies (K=46, stride 48) ===============================
  for (int i = tid; i < 12288; i += 256) sX[i] = (h16)0.f;
  __syncthreads();
  if (tid < 240) {
    int ent = tid >> 4, b = tid & 15;
    h16* dst = sX + (size_t)tid * 48;
    const float* src = p.enemies + ((size_t)(b0 + b) * 15 + ent) * 38;
    float mabs = 0.f;
    for (int j = 0; j < 38; ++j) { float x = src[j]; dst[j] = (h16)x; mabs += fabsf(x); }
    int am = 0; float best = src[29];
    for (int k = 1; k < 9; ++k) { float x = src[29 + k]; if (x > best) { best = x; am = k; } }
    for (int k = 0; k < 8; ++k) dst[38 + k] = (h16)p.type_emb[am * 8 + k];
    sMask[224 + ent * 16 + b] = (mabs > 0.f) ? 1.f : 0.f;
  }
  __syncthreads();
  for (int e = wave; e < 15; e += 8) {
    v8f acc[12] = {};
    gemm_rowtile<12>(sX + (size_t)e * 16 * 48, 48, wp, p.off[M_EN1], 2, lane, acc);
    h16* dst = sEN + (size_t)e * 3072;
#pragma unroll
    for (int t = 0; t < 12; ++t) store_tile_f16_gelu(dst, 192, t, acc[t], lane, p.b_en1);
  }
  __syncthreads();
  ln_rows_f16(sEN, 240, 192, 192, p.g_en1, p.bt_en1, wave, lane);
  __syncthreads();
  for (int e = wave; e < 15; e += 8) {
    h16* buf = sEN + (size_t)e * 3072;
    v8f acc[12] = {};
    gemm_rowtile<12>(buf, 192, wp, p.off[M_EN2], 6, lane, acc);
#pragma unroll
    for (int t = 0; t < 12; ++t) store_tile_f16(buf, 192, t, acc[t], lane, p.b_en2);
  }
  __syncthreads();

  // ===== Phase A: cross-attention (ally then enemy) ========================
  float* qBuf = fA;            // 16x192
  float* oBuf = fA + 3072;     // 16x192
  const float rscale = 0.14433756729740643f;   // 1/sqrt(48)
  for (int side = 0; side < 2; ++side) {
    const h16* src = side ? sEN : sAL;
    const int M = side ? 15 : 14;
    const unsigned oq = p.off[side ? M_EQ : M_AQ], ok = p.off[side ? M_EK : M_AK];
    const unsigned ov = p.off[side ? M_EV : M_AV], oo = p.off[side ? M_EO : M_AO];
    const float *bq = side ? p.bq_e : p.bq_a, *bk = side ? p.bk_e : p.bk_a;
    const float *bv = side ? p.bv_e : p.bv_a, *bo = side ? p.bo_e : p.bo_a;
    const float* mk = sMask + (side ? 224 : 0);
    float* cDst = side ? (fB + 3072) : fB;

    // q = self_emb @ Wq + bq
    for (int nt = wave; nt < 12; nt += 8) {
      v8f c = gemm_one_tile(sSELF, 192, wp, oq, 6, nt, lane);
      store_tile_f32(qBuf, 192, nt, c, lane, bq);
    }
    __syncthreads();

    // scores: per wave, per entity, K_m kept in fragments; reduce to s[b][h][m].
    for (int m = wave; m < M; m += 8) {
      v8f acc[12] = {};
      gemm_rowtile<12>(src + (size_t)m * 3072, 192, wp, ok, 6, lane, acc);
#pragma unroll
      for (int hh = 0; hh < 4; ++hh) {
        float pv[8] = {0,0,0,0,0,0,0,0};
#pragma unroll
        for (int t = 0; t < 3; ++t) {
          int nt = 3 * hh + t, col = nt * 16 + cl;
          float bkv = bk[col];
#pragma unroll
          for (int v = 0; v < 8; ++v)
            pv[v] += (acc[nt][v] + bkv) * qBuf[(size_t)(v + 8 * kh) * 192 + col];
        }
#pragma unroll
        for (int v = 0; v < 8; ++v) {
          float x = pv[v];
#pragma unroll
          for (int o = 8; o > 0; o >>= 1) x += __shfl_xor(x, o, 32);
          if (cl == 0) sSc[(((v + 8 * kh) * 4 + hh) * 16) + m] = x * rscale;
        }
      }
    }
    __syncthreads();

    // masked softmax over m for each (b,h)
    if (tid < 64) {
      int b = tid >> 2, hh = tid & 3;
      float vbuf[16]; float mx = -1e30f;
      for (int m = 0; m < M; ++m) {
        float s = sSc[((b * 4 + hh) * 16) + m];
        s = (mk[m * 16 + b] > 0.f) ? s : -1e9f;
        vbuf[m] = s; mx = fmaxf(mx, s);
      }
      float den = 0.f;
      for (int m = 0; m < M; ++m) { float e = expf(vbuf[m] - mx); vbuf[m] = e; den += e; }
      float rr = 1.f / den;
      for (int m = 0; m < M; ++m) sSc[((b * 4 + hh) * 16) + m] = vbuf[m] * rr;
    }
    // zero o accumulator
    for (int i = tid; i < 3072; i += 256) oBuf[i] = 0.f;
    __syncthreads();

    // o += p[b][h][m] * (V_m + bv)
    for (int m = wave; m < M; m += 8) {
      v8f acc[12] = {};
      gemm_rowtile<12>(src + (size_t)m * 3072, 192, wp, ov, 6, lane, acc);
#pragma unroll
      for (int nt = 0; nt < 12; ++nt) {
        int col = nt * 16 + cl;
        float bvv = bv[col];
#pragma unroll
        for (int v = 0; v < 8; ++v) {
          int b = v + 8 * kh;
          float pr = sSc[((b * 4 + (nt / 3)) * 16) + m];
          atomicAdd(&oBuf[(size_t)b * 192 + col], pr * (acc[nt][v] + bvv));
        }
      }
    }
    __syncthreads();
    // out-proj: o @ Wo + bo -> cA / cE
    for (int i = tid; i < 3072; i += 256) sX[i] = (h16)oBuf[i];
    __syncthreads();
    for (int nt = wave; nt < 12; nt += 8) {
      v8f c = gemm_one_tile(sX, 192, wp, oo, 6, nt, lane);
      store_tile_f32(cDst, 192, nt, c, lane, bo);
    }
    __syncthreads();
  }
  float* cA = fB;         // 16x192
  float* cE = fB + 3072;  // 16x192
  float* h0 = fB + 6144;  // 16x192

  // ===== Phase F: fuse MLP  x=[self,cA,cE] -> h0 ===========================
  for (int i = tid; i < 9216; i += 256) {
    int b = i / 576, j = i % 576;
    float v = (j < 192) ? (float)sSELF[(size_t)b * 192 + j]
            : (j < 384) ? cA[(size_t)b * 192 + (j - 192)]
                        : cE[(size_t)b * 192 + (j - 384)];
    sX[i] = (h16)v;
  }
  __syncthreads();
  for (int nt = wave; nt < 12; nt += 8) {
    v8f c = gemm_one_tile(sX, 576, wp, p.off[M_FU1], 18, nt, lane);
    store_tile_f16_gelu(sX2, 192, nt, c, lane, p.b_fu1);
  }
  __syncthreads();
  ln_rows_f16(sX2, 16, 192, 192, p.g_fu1, p.bt_fu1, wave, lane);
  __syncthreads();
  for (int nt = wave; nt < 12; nt += 8) {
    v8f c = gemm_one_tile(sX2, 192, wp, p.off[M_FU2], 6, nt, lane);
    store_tile_f32(h0, 192, nt, c, lane, p.b_fu2);
  }
  __syncthreads();

  // ===== Phase P: masked-mean pooling + pool/role MLPs =====================
  for (int i = tid; i < 3072; i += 256) {
    int b = i / 192, j = i % 192;
    float sa = 0.f, wa = 0.f, se = 0.f, we = 0.f;
    for (int e = 0; e < 14; ++e) {
      float m = sMask[e * 16 + b]; wa += m;
      sa += m * (float)sAL[(size_t)e * 3072 + (size_t)b * 192 + j];
    }
    for (int e = 0; e < 15; ++e) {
      float m = sMask[224 + e * 16 + b]; we += m;
      se += m * (float)sEN[(size_t)e * 3072 + (size_t)b * 192 + j];
    }
    fA[i]        = sa / fmaxf(wa, 1e-8f);  // pA
    fA[3072 + i] = se / fmaxf(we, 1e-8f);  // pE
  }
  __syncthreads();
  for (int i = tid; i < 9216; i += 256) {
    int b = i / 576, j = i % 576;
    float v = (j < 192) ? (float)sSELF[(size_t)b * 192 + j]
            : (j < 384) ? fA[(size_t)b * 192 + (j - 192)]
                        : fA[3072 + (size_t)b * 192 + (j - 384)];
    sX[i] = (h16)v;
  }
  __syncthreads();
  for (int nt = wave; nt < 8; nt += 8) {
    v8f c = gemm_one_tile(sX, 576, wp, p.off[M_PO1], 18, nt, lane);
    store_tile_f16_gelu(sX2, 128, nt, c, lane, p.b_po1);
  }
  for (int nt = wave + 8; nt < 8; nt += 8) {}  // (8 tiles: waves 0..7 each take one)
  // NOTE: loop above already covers nt = wave for wave<8.
  __syncthreads();
  ln_rows_f16(sX2, 16, 128, 128, p.g_po1, p.bt_po1, wave, lane);
  __syncthreads();
  float* pooled = fA + 6144;  // 16x128
  for (int nt = wave; nt < 8; nt += 8) {
    v8f c = gemm_one_tile(sX2, 128, wp, p.off[M_PO2], 4, nt, lane);
    store_tile_f32(pooled, 128, nt, c, lane, p.b_po2);
  }
  __syncthreads();
  // role MLP
  for (int i = tid; i < 2048; i += 256) {
    int b = i >> 7, j = i & 127;
    sX[i] = (h16)pooled[(size_t)b * 128 + j];
  }
  __syncthreads();
  for (int nt = wave; nt < 8; nt += 8) {
    v8f c = gemm_one_tile(sX, 128, wp, p.off[M_RO1], 4, nt, lane);
    store_tile_f16_gelu(sX2, 128, nt, c, lane, p.b_ro1);
  }
  __syncthreads();
  ln_rows_f16(sX2, 16, 128, 128, p.g_ro1, p.bt_ro1, wave, lane);
  __syncthreads();
  float* rlog = fA + 8192;  // 16x4
  if (wave == 0) {
    v8f c = gemm_one_tile(sX2, 128, wp, p.off[M_RO2], 4, 0, lane);
    if (cl < 4) {
#pragma unroll
      for (int v = 0; v < 8; ++v)
        rlog[(size_t)(v + 8 * kh) * 4 + cl] = c[v] + p.b_ro2[cl];
    }
  }
  __syncthreads();
  if (tid < 16) {
    int b = tid;
    float m0 = fmaxf(fmaxf(rlog[b*4+0], rlog[b*4+1]), fmaxf(rlog[b*4+2], rlog[b*4+3]));
    float e0 = expf(rlog[b*4+0]-m0), e1 = expf(rlog[b*4+1]-m0);
    float e2 = expf(rlog[b*4+2]-m0), e3 = expf(rlog[b*4+3]-m0);
    float inv = 1.f / (e0+e1+e2+e3);
    sR[b*4+0]=e0*inv; sR[b*4+1]=e1*inv; sR[b*4+2]=e2*inv; sR[b*4+3]=e3*inv;
  }
  __syncthreads();

  // ===== Phase G: GRU ======================================================
  // x = LN(concat(h0, r)) with pre_g/pre_b, width 196, staged f16 stride 224.
  for (int i = tid; i < 3584; i += 256) sX[i] = (h16)0.f;
  // stage f16 hidden (stride 256) while zeroing
  for (int i = tid; i < 4096; i += 256) {
    int b = i >> 8, j = i & 255;
    sHID[i] = (h16)p.hidden[(size_t)(b0 + b) * 256 + j];
  }
  __syncthreads();
  for (int r = wave; r < 16; r += 8) {
    float s = 0.f, s2 = 0.f;
    for (int j = lane; j < 196; j += 32) {
      float x = (j < 192) ? h0[(size_t)r * 192 + j] : sR[r * 4 + (j - 192)];
      s += x; s2 += x * x;
    }
#pragma unroll
    for (int o = 16; o > 0; o >>= 1) { s += __shfl_xor(s, o, 32); s2 += __shfl_xor(s2, o, 32); }
    float m = s / 196.f, vv = s2 / 196.f - m * m;
    float inv = rsqrtf(vv + 1e-5f);
    for (int j = lane; j < 196; j += 32) {
      float x = (j < 192) ? h0[(size_t)r * 192 + j] : sR[r * 4 + (j - 192)];
      sX[(size_t)r * 224 + j] = (h16)(((x - m) * inv) * p.pre_g[j] + p.pre_b[j]);
    }
  }
  __syncthreads();
  // gi = x @ Wih + bih (48 tiles), gh = h @ Whh + bhh (48 tiles)
  for (int nt = wave; nt < 48; nt += 8) {
    v8f c = gemm_one_tile(sX, 224, wp, p.off[M_WIH], 7, nt, lane);
    store_tile_f32(fA, 768, nt, c, lane, p.gbih);
  }
  for (int nt = wave; nt < 48; nt += 8) {
    v8f c = gemm_one_tile(sHID, 256, wp, p.off[M_WHH], 8, nt, lane);
    store_tile_f32(fB, 768, nt, c, lane, p.gbhh);
  }
  __syncthreads();
  // gates -> new_h (global) + f16 copy for LN/trunk
  for (int i = tid; i < 4096; i += 256) {
    int b = i >> 8, j = i & 255;
    float ir = fA[(size_t)b * 768 + j],       iz = fA[(size_t)b * 768 + 256 + j];
    float in_ = fA[(size_t)b * 768 + 512 + j];
    float hr = fB[(size_t)b * 768 + j],       hz = fB[(size_t)b * 768 + 256 + j];
    float hn = fB[(size_t)b * 768 + 512 + j];
    float rg = sigmoidf_(ir + hr);
    float zg = sigmoidf_(iz + hz);
    float ng = tanhf(in_ + rg * hn);
    float h  = p.hidden[(size_t)(b0 + b) * 256 + j];
    float nh = (1.f - zg) * ng + zg * h;
    p.outH[(size_t)(b0 + b) * 256 + j] = nh;
    sX[i] = (h16)nh;                       // 16x256, stride 256
  }
  __syncthreads();
  ln_rows_f16(sX, 16, 256, 256, p.post_g, p.post_b, wave, lane);
  __syncthreads();
  // trunk MLP: 256 -> 192 -> 192
  for (int nt = wave; nt < 12; nt += 8) {
    v8f c = gemm_one_tile(sX, 256, wp, p.off[M_TR1], 8, nt, lane);
    store_tile_f16_gelu(sX2, 192, nt, c, lane, p.b_tr1);
  }
  __syncthreads();
  ln_rows_f16(sX2, 16, 192, 192, p.g_tr1, p.bt_tr1, wave, lane);
  __syncthreads();
  float* zBuf = fB;  // 16x192 (gh dead)
  for (int nt = wave; nt < 12; nt += 8) {
    v8f c = gemm_one_tile(sX2, 192, wp, p.off[M_TR2], 6, nt, lane);
    store_tile_f32(zBuf, 192, nt, c, lane, p.b_tr2);
  }
  __syncthreads();

  // ===== Phase Q: heads ====================================================
  // zs = concat(z, r), staged f16 stride 224 (zero pad 196..223).
  for (int i = tid; i < 3584; i += 256) {
    int b = i / 224, j = i % 224;
    float v = (j < 192) ? zBuf[(size_t)b * 192 + j]
            : (j < 196) ? sR[b * 4 + (j - 192)] : 0.f;
    sX[i] = (h16)v;
  }
  __syncthreads();
  // Q_move raw (tiny: VALU dot)
  if (tid < 96) {
    int b = tid / 6, j = tid % 6;
    float s = p.mv_b[j];
    for (int k = 0; k < 196; ++k)
      s += (float)sX[(size_t)b * 224 + k] * p.mv_W[(size_t)k * 6 + j];
    sQM[b * 6 + j] = s;
  }
  // q = zs @ qp_W
  for (int nt = wave; nt < 12; nt += 8) {
    v8f c = gemm_one_tile(sX, 224, wp, p.off[M_QP], 7, nt, lane);
    store_tile_f32(fA, 192, nt, c, lane, (const float*)0);
  }
  __syncthreads();
  // logits[b][m] = q . (enemy_emb[m] @ kp) / sqrt(192)
  const float rs192 = 0.07216878364870322f;
  for (int m = wave; m < 15; m += 8) {
    v8f acc[12] = {};
    gemm_rowtile<12>(sEN + (size_t)m * 3072, 192, wp, p.off[M_KP], 6, lane, acc);
    float pv[8] = {0,0,0,0,0,0,0,0};
#pragma unroll
    for (int nt = 0; nt < 12; ++nt) {
      int col = nt * 16 + cl;
#pragma unroll
      for (int v = 0; v < 8; ++v)
        pv[v] += acc[nt][v] * fA[(size_t)(v + 8 * kh) * 192 + col];
    }
#pragma unroll
    for (int v = 0; v < 8; ++v) {
      float x = pv[v];
#pragma unroll
      for (int o = 8; o > 0; o >>= 1) x += __shfl_xor(x, o, 32);
      if (cl == 0) sSc[(v + 8 * kh) * 16 + m] = x * rs192;
    }
  }
  __syncthreads();
  // final assembly of Q = [Q_move (6), Q_shoot (15)]
  if (tid < 16) {
    int b = tid;
    float qm[6]; float ms = 0.f;
    for (int j = 0; j < 6; ++j) { qm[j] = sQM[b * 6 + j]; ms += qm[j]; }
    ms *= (1.f / 6.f);
    float* oq = p.outQ + (size_t)(b0 + b) * 21;
    for (int j = 0; j < 6; ++j) oq[j] = qm[j] - ms;
    float lg[15]; float dm = 0.f, sv = 0.f;
    for (int m = 0; m < 15; ++m) {
      float mk = sMask[224 + m * 16 + b];
      float l = (mk > 0.f) ? sSc[b * 16 + m] : -1e9f;
      lg[m] = l;
      if (mk > 0.f) { sv += l; dm += 1.f; }
    }
    float mv = sv / fmaxf(dm, 1.f);
    for (int m = 0; m < 15; ++m) oq[6 + m] = lg[m] - mv;
  }
}

// ---------------------------------------------------------------------------
// Host side.
// d_in flatten order (setup_inputs dict order, params leaves in insertion
// order): 0 own, 1 allies, 2 enemies, 3 hidden, 4 type_emb,
// 5..10 enc_self{W1,b1,g1,bt1,W2,b2}, 11..16 enc_ally, 17..22 enc_enemy,
// 23..30 attn_ally{Wq,bq,Wk,bk,Wv,bv,Wo,bo}, 31..38 attn_enemy,
// 39..44 fuse, 45..50 pool_proj, 51..56 role, 57 pre_g, 58 pre_b,
// 59 gru_Wih, 60 gru_Whh, 61 gru_bih, 62 gru_bhh, 63 post_g, 64 post_b,
// 65..70 trunk, 71 mv_W, 72 mv_b, 73 qp_W, 74 kp_W.
// ---------------------------------------------------------------------------
extern "C" void kernel_launch(void* const* d_in, const int* in_sizes, int n_in,
                              void* d_out, int out_size, void* d_ws, size_t ws_size,
                              hipStream_t stream) {
  if (n_in < 75) return;
  auto F = [&](int i) { return (const float*)d_in[i]; };

  static const int srcIdx[NMAT] = {5,9, 11,15, 17,21, 23,25,27,29, 31,33,35,37,
                                   39,43, 45,49, 51,55, 59,60, 65,69, 73,74};
  static const int MK[NMAT] = {40,192, 48,192, 46,192, 192,192,192,192,
                               192,192,192,192, 576,192, 576,128, 128,128,
                               196,256, 256,192, 196,192};
  static const int MN[NMAT] = {192,192, 192,192, 192,192, 192,192,192,192,
                               192,192,192,192, 192,192, 128,128, 128,4,
                               768,768, 192,192, 192,192};

  PackArgs pa;
  unsigned cum = 0, off = 0;
  for (int i = 0; i < NMAT; ++i) {
    pa.src[i] = F(srcIdx[i]);
    pa.K[i] = MK[i]; pa.N[i] = MN[i];
    pa.off[i] = off; pa.cum[i] = cum;
    unsigned KB = (unsigned)(MK[i] + 31) / 32, NT = (unsigned)(MN[i] + 15) / 16;
    cum += KB * NT;
    off += KB * NT * 512u;
  }
  pa.cum[NMAT] = cum;
  pa.dst = (h16*)d_ws;
  if (ws_size < (size_t)off * sizeof(h16)) return;

  int packBlocks = (int)((cum + 7) / 8);
  capernn_pack_weights<<<packBlocks, 256, 0, stream>>>(pa);

  KParams kp;
  kp.own = F(0); kp.allies = F(1); kp.enemies = F(2); kp.hidden = F(3); kp.type_emb = F(4);
  kp.b_self1 = F(6);  kp.g_self1 = F(7);  kp.bt_self1 = F(8);  kp.b_self2 = F(10);
  kp.b_ally1 = F(12); kp.g_ally1 = F(13); kp.bt_ally1 = F(14); kp.b_ally2 = F(16);
  kp.b_en1   = F(18); kp.g_en1   = F(19); kp.bt_en1   = F(20); kp.b_en2   = F(22);
  kp.bq_a = F(24); kp.bk_a = F(26); kp.bv_a = F(28); kp.bo_a = F(30);
  kp.bq_e = F(32); kp.bk_e = F(34); kp.bv_e = F(36); kp.bo_e = F(38);
  kp.b_fu1 = F(40); kp.g_fu1 = F(41); kp.bt_fu1 = F(42); kp.b_fu2 = F(44);
  kp.b_po1 = F(46); kp.g_po1 = F(47); kp.bt_po1 = F(48); kp.b_po2 = F(50);
  kp.b_ro1 = F(52); kp.g_ro1 = F(53); kp.bt_ro1 = F(54); kp.b_ro2 = F(56);
  kp.pre_g = F(57); kp.pre_b = F(58);
  kp.gbih = F(61); kp.gbhh = F(62);
  kp.post_g = F(63); kp.post_b = F(64);
  kp.b_tr1 = F(66); kp.g_tr1 = F(67); kp.bt_tr1 = F(68); kp.b_tr2 = F(70);
  kp.mv_W = F(71); kp.mv_b = F(72);
  kp.wp = (const h16*)d_ws;
  for (int i = 0; i < NMAT; ++i) kp.off[i] = pa.off[i];
  kp.outQ = (float*)d_out;
  kp.outH = kp.outQ + (size_t)BN_TOT * 21;

  (void)in_sizes; (void)out_size;
  hipFuncSetAttribute((const void*)capernn_fused,
                      hipFuncAttributeMaxDynamicSharedMemorySize, SMEM_BYTES);
  capernn_fused<<<dim3(BN_TOT / 16), dim3(256), SMEM_BYTES, stream>>>(kp);
}
